// PointTransformerLayer_38268158607587
// MI455X (gfx1250) — matrix operand order repaired and checked by reading, not compiled, not run on hardware
//
#include <hip/hip_runtime.h>
#include <hip/hip_fp16.h>

#define NPTS   32768
#define BATCH  8
#define NPB    4096
#define KNN    16
#define FEAT   64
#define HD     8
#define EPS    1e-5f
// LDS distance tile: lane-major permuted layout.
// column c of a row lives at (c&31)*CSTRIDE + (c>>5); CSTRIDE=132 (128 + 4 pad)
// row stride DROW = 32*132 = 4224 floats; 16 rows * 4224 * 4B = 270336 B < 320KB
#define CSTRIDE 132
#define DROW    (32 * CSTRIDE)

typedef __attribute__((ext_vector_type(16))) _Float16 v16h;
typedef __attribute__((ext_vector_type(8)))  _Float16 v8h;
typedef __attribute__((ext_vector_type(8)))  float    v8f;
typedef __attribute__((ext_vector_type(4)))  float    v4f;

// ---------------- WMMA operand loaders (CDNA5 16-bit layouts, ISA 7.12.2) ----
// A 16x32 f16: lane m (0-15): elems0-7 = K k0+0..7, elems8-15 = K k0+16..23
//              lane m+16   : elems0-7 = K k0+8..15, elems8-15 = K k0+24..31
static __device__ inline v16h load_wmma_a(const _Float16* base, int ld, int row0, int k0) {
  int lane = threadIdx.x & 31;
  int m = lane & 15, hi = lane >> 4;
  const _Float16* rp = base + (size_t)(row0 + m) * ld + k0 + hi * 8;
  v8h lo = *reinterpret_cast<const v8h*>(rp);        // 16B aligned
  v8h hh = *reinterpret_cast<const v8h*>(rp + 16);
  v16h a;
#pragma unroll
  for (int e = 0; e < 8; ++e) { a[e] = lo[e]; a[8 + e] = hh[e]; }
  return a;
}
// B 32x16 f16 where B[k][j] = mat[col0+j][k] (mat row-major, row = one column of B):
// lane j (0-15): elems0-7 = K k0+0..7, elems8-15 = K k0+16..23 ; lane j+16: K k0+8..15, k0+24..31
static __device__ inline v16h load_wmma_b(const _Float16* base, int ld, int col0, int k0) {
  int lane = threadIdx.x & 31;
  int j = lane & 15, hi = lane >> 4;
  const _Float16* cp = base + (size_t)(col0 + j) * ld + k0 + hi * 8;
  v8h lo = *reinterpret_cast<const v8h*>(cp);        // 16B aligned
  v8h hh = *reinterpret_cast<const v8h*>(cp + 16);
  v16h b;
#pragma unroll
  for (int e = 0; e < 8; ++e) { b[e] = lo[e]; b[8 + e] = hh[e]; }
  return b;
}

static __device__ inline float waveSum(float v) {
#pragma unroll
  for (int o = 16; o; o >>= 1) v += __shfl_xor(v, o, 32);
  return v;
}

// ---------------- prep kernels ----------------------------------------------
__global__ void cvt_f16_kernel(const float* in, _Float16* out, int n) {
  int i = blockIdx.x * blockDim.x + threadIdx.x;
  if (i < n) out[i] = (_Float16)in[i];
}

// W[64,64] row-major -> Wt_h[j][k] = (f16) W[k][j]
__global__ void cvt_wt_kernel(const float* wsrc, _Float16* wt) {
  int i = blockIdx.x * blockDim.x + threadIdx.x;   // 4096
  int j = i >> 6, k = i & 63;
  wt[i] = (_Float16)wsrc[k * FEAT + j];
}

// ---------------- projection GEMM: C = x @ W + b, via WMMA -------------------
__global__ void __launch_bounds__(32)
proj_kernel(const _Float16* xh, const _Float16* wt, const float* bias, float* outm) {
  int row0 = blockIdx.x * 16, col0 = blockIdx.y * 16;
  v8f acc = {};
#pragma unroll
  for (int k0 = 0; k0 < FEAT; k0 += 32) {
    v16h a = load_wmma_a(xh, FEAT, row0, k0);
    v16h b = load_wmma_b(wt, FEAT, col0, k0);
    acc = __builtin_amdgcn_wmma_f32_16x16x32_f16(false, a, false, b, (short)0, acc, false, false);
  }
  int lane = threadIdx.x & 31;
  int nn = lane & 15, hi = lane >> 4;
  float bb = bias[col0 + nn];
#pragma unroll
  for (int rr = 0; rr < 8; ++rr)
    outm[(size_t)(row0 + rr + hi * 8) * FEAT + col0 + nn] = acc[rr] + bb;
}

// f16 copies + squared row norms for xk / xv
__global__ void post_proj_kernel(const float* xk, const float* xv,
                                 _Float16* xkh, _Float16* xvh, float* sqk, float* sqv) {
  int n = blockIdx.x * blockDim.x + threadIdx.x;
  float sk = 0.f, sv = 0.f;
#pragma unroll 4
  for (int c = 0; c < FEAT; ++c) {
    float a = xk[(size_t)n * FEAT + c];
    float b = xv[(size_t)n * FEAT + c];
    xkh[(size_t)n * FEAT + c] = (_Float16)a;
    xvh[(size_t)n * FEAT + c] = (_Float16)b;
    sk += a * a;  sv += b * b;
  }
  sqk[n] = sk;  sqv[n] = sv;
}

// ---------------- KNN: WMMA distance tile in LDS + single-scan top-16 -------
__global__ void __launch_bounds__(256)
knn_kernel(const _Float16* fh, const float* sq, int* idx_out) {
  extern __shared__ float dist[];                 // 16 rows * DROW floats
  int b = blockIdx.y;
  int row0 = blockIdx.x * 16;                     // local row base within batch
  const _Float16* fb = fh + (size_t)b * NPB * FEAT;
  const float* sqb = sq + b * NPB;
  int lane = threadIdx.x & 31;
  int wave = threadIdx.x >> 5;
  int nn = lane & 15, hi = lane >> 4;

  v16h a0 = load_wmma_a(fb, FEAT, row0, 0);       // A tile is loop-invariant
  v16h a1 = load_wmma_a(fb, FEAT, row0, 32);
  float sqrow[8];
#pragma unroll
  for (int rr = 0; rr < 8; ++rr) sqrow[rr] = sqb[row0 + rr + hi * 8];

  for (int jt = wave; jt < NPB / 16; jt += 8) {   // 8 waves stripe 256 col tiles
    int col0 = jt * 16;
    v16h b0 = load_wmma_b(fb, FEAT, col0, 0);
    v16h b1 = load_wmma_b(fb, FEAT, col0, 32);
    v8f acc = {};
    acc = __builtin_amdgcn_wmma_f32_16x16x32_f16(false, a0, false, b0, (short)0, acc, false, false);
    acc = __builtin_amdgcn_wmma_f32_16x16x32_f16(false, a1, false, b1, (short)0, acc, false, false);
    float sqc = sqb[col0 + nn];
    // permuted store: column c -> (c&31)*CSTRIDE + (c>>5); no carry since col0%16==0, nn<16
    int off = ((col0 & 31) + nn) * CSTRIDE + (col0 >> 5);
#pragma unroll
    for (int rr = 0; rr < 8; ++rr) {
      int m = rr + hi * 8;                        // C layout: VGPR rr, M = rr + 8*hi
      dist[m * DROW + off] = sqrow[rr] + sqc - 2.0f * acc[rr];
    }
  }
  __syncthreads();

  // single-scan top-16: lane's 128 columns are contiguous (b128 loads);
  // 32-bit keys = 20 monotonic-float bits | 12-bit column (exact index tie-break)
  for (int rr2 = 0; rr2 < 2; ++rr2) {
    int row = wave * 2 + rr2;
    const float* rp = dist + row * DROW + lane * CSTRIDE;
    unsigned lst[16];
#pragma unroll
    for (int i = 0; i < 16; ++i) lst[i] = 0xFFFFFFFFu;
#pragma unroll 4
    for (int j4 = 0; j4 < 128; j4 += 4) {
      v4f d4 = *reinterpret_cast<const v4f*>(rp + j4);
#pragma unroll
      for (int e = 0; e < 4; ++e) {
        int c = (j4 + e) * 32 + lane;             // inverse of the permutation
        unsigned u = __float_as_uint(d4[e]);
        u = (u & 0x80000000u) ? ~u : (u | 0x80000000u);
        unsigned key = (u & 0xFFFFF000u) | (unsigned)c;
        if (key < lst[15]) {                      // guarded sorted-insert, static idx
          unsigned x = key;
#pragma unroll
          for (int i = 0; i < 16; ++i) {
            unsigned lo = lst[i] < x ? lst[i] : x;
            unsigned hh = lst[i] < x ? x : lst[i];
            lst[i] = lo;  x = hh;
          }
        }
      }
    }
    // merge 32 sorted lists: 16 rounds of wave-min; unique winner pops its head
    for (int t = 0; t < KNN; ++t) {
      unsigned best = lst[0];
#pragma unroll
      for (int o = 16; o; o >>= 1) {
        unsigned oth = __shfl_down(best, o, 32);
        if (oth < best) best = oth;
      }
      best = __shfl(best, 0, 32);
      if (lane == 0)
        idx_out[(size_t)(b * NPB + row0 + row) * KNN + t] = b * NPB + (int)(best & 0xFFFu);
      if (lst[0] == best) {                       // keys unique (column embedded)
#pragma unroll
        for (int i = 0; i < 15; ++i) lst[i] = lst[i + 1];
        lst[15] = 0xFFFFFFFFu;
      }
    }
  }
}

// ---------------- recompute helpers (xq/xk L2-resident -> recompute is cheap) -
static __device__ inline void compute_pr_r(
    int n, int t, const float* p, const int* idxk, const float* xq, const float* xk,
    const float* wp1, const float* bp1, const float* scp, const float* shp,
    const float* wp2, const float* bp2, float* p_r, float* r) {
  int j = idxk[(size_t)n * KNN + t];
  float pr0 = p[3 * j + 0] - p[3 * n + 0];
  float pr1 = p[3 * j + 1] - p[3 * n + 1];
  float pr2 = p[3 * j + 2] - p[3 * n + 2];
  float h[3];
#pragma unroll
  for (int c = 0; c < 3; ++c) {
    float v = bp1[c] + pr0 * wp1[c] + pr1 * wp1[3 + c] + pr2 * wp1[6 + c];
    v = v * scp[c] + shp[c];
    h[c] = v > 0.f ? v : 0.f;
  }
  const float* xqn = xq + (size_t)n * FEAT;
  const float* xkj = xk + (size_t)j * FEAT;
#pragma unroll
  for (int c = 0; c < FEAT; ++c) {
    float v = bp2[c] + h[0] * wp2[c] + h[1] * wp2[FEAT + c] + h[2] * wp2[2 * FEAT + c];
    p_r[c] = v;
    r[c] = xkj[c] - xqn[c] + v;
  }
}

static __device__ inline void compute_w1(const float* r, const float* sc1, const float* sh1,
                                         const float* ww1, const float* bw1, float* w1) {
#pragma unroll
  for (int i = 0; i < HD; ++i) w1[i] = bw1[i];
  for (int c = 0; c < FEAT; ++c) {
    float a = r[c] * sc1[c] + sh1[c];  a = a > 0.f ? a : 0.f;
#pragma unroll
    for (int i = 0; i < HD; ++i) w1[i] += a * ww1[c * HD + i];
  }
}

static __device__ inline void compute_w2(const float* w1, const float* sc2, const float* sh2,
                                         const float* ww2, const float* bw2, float* wo) {
#pragma unroll
  for (int i = 0; i < HD; ++i) wo[i] = bw2[i];
#pragma unroll
  for (int j = 0; j < HD; ++j) {
    float a = w1[j] * sc2[j] + sh2[j];  a = a > 0.f ? a : 0.f;
#pragma unroll
    for (int i = 0; i < HD; ++i) wo[i] += a * ww2[j * HD + i];
  }
}

// ---------------- global batch-norm stats passes ----------------------------
__global__ void stats_h_kernel(const float* p, const int* idxk,
                               const float* wp1, const float* bp1,
                               float* sum_h, float* sq_h) {
  __shared__ float s[6];
  if (threadIdx.x < 6) s[threadIdx.x] = 0.f;
  __syncthreads();
  size_t gid = (size_t)blockIdx.x * blockDim.x + threadIdx.x;   // over N*K exactly
  int n = (int)(gid / KNN);
  int j = idxk[gid];
  float pr0 = p[3 * j + 0] - p[3 * n + 0];
  float pr1 = p[3 * j + 1] - p[3 * n + 1];
  float pr2 = p[3 * j + 2] - p[3 * n + 2];
#pragma unroll
  for (int c = 0; c < 3; ++c) {
    float h = bp1[c] + pr0 * wp1[c] + pr1 * wp1[3 + c] + pr2 * wp1[6 + c];
    float sv = waveSum(h), qv = waveSum(h * h);
    if ((threadIdx.x & 31) == 0) { atomicAdd(&s[c], sv); atomicAdd(&s[3 + c], qv); }
  }
  __syncthreads();
  if (threadIdx.x < 3)      atomicAdd(&sum_h[threadIdx.x], s[threadIdx.x]);
  else if (threadIdx.x < 6) atomicAdd(&sq_h[threadIdx.x - 3], s[threadIdx.x]);
}

__global__ void __launch_bounds__(256)
stats_r_kernel(const float* p, const int* idxk, const float* xq, const float* xk,
               const float* wp1, const float* bp1, const float* scp, const float* shp,
               const float* wp2, const float* bp2, float* sum_r, float* sq_r) {
  __shared__ float ss[128];
  if (threadIdx.x < 128) ss[threadIdx.x] = 0.f;
  __syncthreads();
  size_t gid = (size_t)blockIdx.x * blockDim.x + threadIdx.x;
  int n = (int)(gid / KNN), t = (int)(gid % KNN);
  float p_r[FEAT], r[FEAT];
  compute_pr_r(n, t, p, idxk, xq, xk, wp1, bp1, scp, shp, wp2, bp2, p_r, r);
  for (int c = 0; c < FEAT; ++c) {
    float s = waveSum(r[c]), q = waveSum(r[c] * r[c]);
    if ((threadIdx.x & 31) == 0) { atomicAdd(&ss[c], s); atomicAdd(&ss[64 + c], q); }
  }
  __syncthreads();
  if (threadIdx.x < 64)       atomicAdd(&sum_r[threadIdx.x], ss[threadIdx.x]);
  else if (threadIdx.x < 128) atomicAdd(&sq_r[threadIdx.x - 64], ss[threadIdx.x]);
}

__global__ void __launch_bounds__(256)
stats_w1_kernel(const float* p, const int* idxk, const float* xq, const float* xk,
                const float* wp1, const float* bp1, const float* scp, const float* shp,
                const float* wp2, const float* bp2,
                const float* sc1, const float* sh1, const float* ww1, const float* bw1,
                float* sum_w, float* sq_w) {
  __shared__ float ss[16];
  if (threadIdx.x < 16) ss[threadIdx.x] = 0.f;
  __syncthreads();
  size_t gid = (size_t)blockIdx.x * blockDim.x + threadIdx.x;
  int n = (int)(gid / KNN), t = (int)(gid % KNN);
  float p_r[FEAT], r[FEAT], w1[HD];
  compute_pr_r(n, t, p, idxk, xq, xk, wp1, bp1, scp, shp, wp2, bp2, p_r, r);
  compute_w1(r, sc1, sh1, ww1, bw1, w1);
#pragma unroll
  for (int i = 0; i < HD; ++i) {
    float s = waveSum(w1[i]), q = waveSum(w1[i] * w1[i]);
    if ((threadIdx.x & 31) == 0) { atomicAdd(&ss[i], s); atomicAdd(&ss[8 + i], q); }
  }
  __syncthreads();
  if (threadIdx.x < 8)       atomicAdd(&sum_w[threadIdx.x], ss[threadIdx.x]);
  else if (threadIdx.x < 16) atomicAdd(&sq_w[threadIdx.x - 8], ss[threadIdx.x]);
}

__global__ void fin_bn_kernel(const float* sum, const float* sumsq, const float* g,
                              const float* b, int nch, float invcnt,
                              float* scale, float* shift) {
  int c = threadIdx.x;
  if (c < nch) {
    float m = sum[c] * invcnt;
    float v = sumsq[c] * invcnt - m * m;
    float sc = g[c] * rsqrtf(v + EPS);
    scale[c] = sc;
    shift[c] = b[c] - m * sc;
  }
}

__global__ void zero_kernel(float* ptr, int n) {
  int i = blockIdx.x * blockDim.x + threadIdx.x;
  if (i < n) ptr[i] = 0.f;
}

// ---------------- fused final: softmax over K + combine ---------------------
__global__ void __launch_bounds__(256)
final_kernel(const float* p, const int* idxk, const int* idxv,
             const float* xq, const float* xk, const float* xv,
             const float* wp1, const float* bp1, const float* scp, const float* shp,
             const float* wp2, const float* bp2,
             const float* sc1, const float* sh1, const float* ww1, const float* bw1,
             const float* sc2, const float* sh2, const float* ww2, const float* bw2,
             float* out) {
  int lane = threadIdx.x & 31;
  int wid  = (int)((blockIdx.x * blockDim.x + threadIdx.x) >> 5);
  int half = lane >> 4;                 // 2 points per wave
  int t    = lane & 15;                 // lane = neighbor index
  int n    = wid * 2 + half;

  float p_r[FEAT], r[FEAT], w1[HD], watt[HD];
  compute_pr_r(n, t, p, idxk, xq, xk, wp1, bp1, scp, shp, wp2, bp2, p_r, r);
  compute_w1(r, sc1, sh1, ww1, bw1, w1);
  compute_w2(w1, sc2, sh2, ww2, bw2, watt);

  // softmax over the 16 neighbors (16-lane group) per hd channel
#pragma unroll
  for (int i = 0; i < HD; ++i) {
    float m = watt[i];
#pragma unroll
    for (int o = 8; o; o >>= 1) m = fmaxf(m, __shfl_xor(m, o, 16));
    float e = __expf(watt[i] - m);
    float s = e;
#pragma unroll
    for (int o = 8; o; o >>= 1) s += __shfl_xor(s, o, 16);
    watt[i] = e / s;
  }

  int jv = idxv[(size_t)n * KNN + t];
  const float* xvj = xv + (size_t)jv * FEAT;
  float* on = out + (size_t)n * FEAT;
#pragma unroll
  for (int c = 0; c < FEAT; ++c) {
    float v = (xvj[c] + p_r[c]) * watt[c & (HD - 1)];   // weight shared across heads
#pragma unroll
    for (int o = 8; o; o >>= 1) v += __shfl_xor(v, o, 16);
    if (t == 0) on[c] = v;
  }
}

// ---------------- launch ----------------------------------------------------
extern "C" void kernel_launch(void* const* d_in, const int* in_sizes, int n_in,
                              void* d_out, int out_size, void* d_ws, size_t ws_size,
                              hipStream_t stream) {
  const float* p   = (const float*)d_in[0];
  const float* x   = (const float*)d_in[1];
  const float* wq  = (const float*)d_in[3];  const float* bq  = (const float*)d_in[4];
  const float* wk  = (const float*)d_in[5];  const float* bk  = (const float*)d_in[6];
  const float* wv  = (const float*)d_in[7];  const float* bv  = (const float*)d_in[8];
  const float* wp1 = (const float*)d_in[9];  const float* bp1 = (const float*)d_in[10];
  const float* gp  = (const float*)d_in[11]; const float* btp = (const float*)d_in[12];
  const float* wp2 = (const float*)d_in[13]; const float* bp2 = (const float*)d_in[14];
  const float* g1  = (const float*)d_in[15]; const float* bt1 = (const float*)d_in[16];
  const float* ww1 = (const float*)d_in[17]; const float* bw1 = (const float*)d_in[18];
  const float* g2  = (const float*)d_in[19]; const float* bt2 = (const float*)d_in[20];
  const float* ww2 = (const float*)d_in[21]; const float* bw2 = (const float*)d_in[22];
  float* out = (float*)d_out;

  char* wsb = (char*)d_ws;  size_t off = 0;
  auto take = [&](size_t bytes) -> void* {
    void* r = wsb + off;  off += (bytes + 255) & ~(size_t)255;  return r;
  };
  float*    xq   = (float*)take((size_t)NPTS * FEAT * 4);
  float*    xk   = (float*)take((size_t)NPTS * FEAT * 4);
  float*    xv   = (float*)take((size_t)NPTS * FEAT * 4);
  _Float16* xh   = (_Float16*)take((size_t)NPTS * FEAT * 2);
  _Float16* xkh  = (_Float16*)take((size_t)NPTS * FEAT * 2);
  _Float16* xvh  = (_Float16*)take((size_t)NPTS * FEAT * 2);
  _Float16* wtq  = (_Float16*)take((size_t)FEAT * FEAT * 2);
  _Float16* wtk  = (_Float16*)take((size_t)FEAT * FEAT * 2);
  _Float16* wtv  = (_Float16*)take((size_t)FEAT * FEAT * 2);
  float*    sqk  = (float*)take((size_t)NPTS * 4);
  float*    sqv  = (float*)take((size_t)NPTS * 4);
  int*      idxk = (int*)take((size_t)NPTS * KNN * 4);
  int*      idxv = (int*)take((size_t)NPTS * KNN * 4);
  float*    st   = (float*)take(512 * 4);
  float *sum_h = st + 0,   *sq_h = st + 8;
  float *sum_r = st + 16,  *sq_r = st + 80;
  float *sum_w = st + 144, *sq_w = st + 152;
  float *scp = st + 160, *shp = st + 168;
  float *sc1 = st + 176, *sh1 = st + 240;
  float *sc2 = st + 304, *sh2 = st + 312;

  const float invcnt = 1.0f / (float)((size_t)NPTS * KNN);

  zero_kernel<<<1, 256, 0, stream>>>(st, 160);

  cvt_f16_kernel<<<(NPTS * FEAT) / 256, 256, 0, stream>>>(x, xh, NPTS * FEAT);
  cvt_wt_kernel<<<16, 256, 0, stream>>>(wq, wtq);
  cvt_wt_kernel<<<16, 256, 0, stream>>>(wk, wtk);
  cvt_wt_kernel<<<16, 256, 0, stream>>>(wv, wtv);

  dim3 pg(NPTS / 16, FEAT / 16);
  proj_kernel<<<pg, 32, 0, stream>>>(xh, wtq, bq, xq);
  proj_kernel<<<pg, 32, 0, stream>>>(xh, wtk, bk, xk);
  proj_kernel<<<pg, 32, 0, stream>>>(xh, wtv, bv, xv);

  post_proj_kernel<<<NPTS / 256, 256, 0, stream>>>(xk, xv, xkh, xvh, sqk, sqv);

  size_t shbytes = (size_t)16 * DROW * 4;      // 270336 B of the 320 KB WGP LDS
  dim3 kg(NPB / 16, BATCH);
  knn_kernel<<<kg, 256, shbytes, stream>>>(xkh, sqk, idxk);
  knn_kernel<<<kg, 256, shbytes, stream>>>(xvh, sqv, idxv);

  int nkBlocks = (NPTS * KNN) / 256;           // 2048, exact
  stats_h_kernel<<<nkBlocks, 256, 0, stream>>>(p, idxk, wp1, bp1, sum_h, sq_h);
  fin_bn_kernel<<<1, 64, 0, stream>>>(sum_h, sq_h, gp, btp, 3, invcnt, scp, shp);

  stats_r_kernel<<<nkBlocks, 256, 0, stream>>>(p, idxk, xq, xk, wp1, bp1, scp, shp,
                                               wp2, bp2, sum_r, sq_r);
  fin_bn_kernel<<<1, 64, 0, stream>>>(sum_r, sq_r, g1, bt1, 64, invcnt, sc1, sh1);

  stats_w1_kernel<<<nkBlocks, 256, 0, stream>>>(p, idxk, xq, xk, wp1, bp1, scp, shp,
                                                wp2, bp2, sc1, sh1, ww1, bw1, sum_w, sq_w);
  fin_bn_kernel<<<1, 64, 0, stream>>>(sum_w, sq_w, g2, bt2, 8, invcnt, sc2, sh2);

  final_kernel<<<NPTS / 16, 256, 0, stream>>>(p, idxk, idxv, xq, xk, xv,
                                              wp1, bp1, scp, shp, wp2, bp2,
                                              sc1, sh1, ww1, bw1, sc2, sh2, ww2, bw2, out);
}
